// MyLSTM_6004364280768
// MI455X (gfx1250) — compile-verified, matrix operand order checked by reading, not run on hardware
//
#include <hip/hip_runtime.h>
#include <stdint.h>

#define DEVI __device__ __forceinline__

typedef __attribute__((ext_vector_type(16))) __bf16 v16bf;
typedef __attribute__((ext_vector_type(8)))  float  v8f;
typedef __attribute__((ext_vector_type(4)))  int    v4i_async;

constexpr int I_SZ = 512, H_SZ = 1024, P_SZ = 1024, BATCH = 32, SEQ = 512;
constexpr int BL = BATCH * SEQ;          // 16384 rows of x
constexpr int NWG_REC = 32;              // persistent workgroups in recurrent kernel
constexpr int HC = H_SZ / NWG_REC;       // 32 H-columns per workgroup

// gfx1250 async global->LDS path (ASYNCcnt), guarded for toolchain support
#if defined(__has_builtin)
#if __has_builtin(__builtin_amdgcn_global_load_async_to_lds_b128) && \
    __has_builtin(__builtin_amdgcn_s_wait_asynccnt)
#define USE_ASYNC_LDS 1
#endif
#endif
#ifndef USE_ASYNC_LDS
#define USE_ASYNC_LDS 0
#endif

// ---------------- bf16 helpers (stored as uint16_t, RNE conversion) --------
DEVI uint16_t f2bf(float x) {
  uint32_t u = __float_as_uint(x);
  uint32_t r = (u + 0x7FFFu + ((u >> 16) & 1u)) >> 16;
  return (uint16_t)r;
}
DEVI float bf2f(uint16_t h) { return __uint_as_float(((uint32_t)h) << 16); }

// ---------------- WMMA fragment load ---------------------------------------
// 16-bit A (16x32) / B-as-W-rows fragment layout per CDNA5 ISA 7.12.2:
// lane l holds row (l&15); lanes 0-15: K = k0+{0..7} and k0+{16..23};
// lanes 16-31: +8.  => two contiguous 16-byte loads per lane.
union FragU { uint4 q[2]; v16bf v; };

DEVI v16bf load_frag(const uint16_t* __restrict__ base, int row, int lda,
                     int k0, int lane) {
  FragU f;
  const uint16_t* p = base + (size_t)row * lda + k0 + ((lane >> 4) << 3);
  f.q[0] = *reinterpret_cast<const uint4*>(p);
  f.q[1] = *reinterpret_cast<const uint4*>(p + 16);
  return f.v;
}

// same, but for LDS-resident tiles (generic pointer rooted in __shared__)
DEVI v16bf load_frag_s(const uint16_t* base, int row, int lda, int k0, int lane) {
  FragU f;
  const uint16_t* p = base + row * lda + k0 + ((lane >> 4) << 3);
  f.q[0] = *reinterpret_cast<const uint4*>(p);
  f.q[1] = *reinterpret_cast<const uint4*>(p + 16);
  return f.v;
}

DEVI v8f wmma_bf16(v16bf a, v16bf b, v8f c) {
  return __builtin_amdgcn_wmma_f32_16x16x32_bf16(
      /*neg_a=*/false, a, /*neg_b=*/false, b,
      /*c_mod=*/(short)0, c, /*reuse_a=*/false, /*reuse_b=*/false);
}

DEVI float sigmoidf_(float x) { return 1.0f / (1.0f + __expf(-x)); }

// ---------------- Kernel: fp32 -> bf16 convert -----------------------------
__global__ void k_convert(const float* __restrict__ src,
                          uint16_t* __restrict__ dst, int n) {
  for (int i = blockIdx.x * blockDim.x + threadIdx.x; i < n;
       i += gridDim.x * blockDim.x)
    dst[i] = f2bf(src[i]);
}

// ---------------- Kernel: init h buffers + barrier counter -----------------
__global__ void k_init(uint16_t* __restrict__ h0, uint16_t* __restrict__ h1,
                       unsigned* __restrict__ cnt, int n) {
  int i = blockIdx.x * blockDim.x + threadIdx.x;
  if (i < n) { h0[i] = 0; h1[i] = 0; }
  if (i == 0) *cnt = 0u;
}

// ---------------- Kernel: input-gate GEMM  G[g] = X @ W[g]^T  (bf16) -------
// grid: (H/64, BL/128, 4 gates), block 256 (8 waves, 4x2 wave tiling, 32x32/wave)
__global__ __launch_bounds__(256) void k_in_gemm(
    const uint16_t* __restrict__ Xb,
    const uint16_t* __restrict__ W0, const uint16_t* __restrict__ W1,
    const uint16_t* __restrict__ W2, const uint16_t* __restrict__ W3,
    uint16_t* __restrict__ G) {
  const int lane = threadIdx.x & 31;
  const int w = threadIdx.x >> 5;          // 0..7
  const int wm = w & 3, wn = w >> 2;       // 4 m-slots x 2 n-slots
  const int g = blockIdx.z;
  const uint16_t* __restrict__ W = (g == 0) ? W0 : (g == 1) ? W1 : (g == 2) ? W2 : W3;
  const int row0 = blockIdx.y * 128 + wm * 32;
  const int col0 = blockIdx.x * 64 + wn * 32;
  const int rr = lane & 15;

  v8f acc[2][2] = {};
#pragma unroll 4
  for (int k0 = 0; k0 < I_SZ; k0 += 32) {
    v16bf a0 = load_frag(Xb, row0 + rr,      I_SZ, k0, lane);
    v16bf a1 = load_frag(Xb, row0 + 16 + rr, I_SZ, k0, lane);
    v16bf b0 = load_frag(W,  col0 + rr,      I_SZ, k0, lane);
    v16bf b1 = load_frag(W,  col0 + 16 + rr, I_SZ, k0, lane);
    acc[0][0] = wmma_bf16(a0, b0, acc[0][0]);
    acc[0][1] = wmma_bf16(a0, b1, acc[0][1]);
    acc[1][0] = wmma_bf16(a1, b0, acc[1][0]);
    acc[1][1] = wmma_bf16(a1, b1, acc[1][1]);
  }
  uint16_t* __restrict__ Gg = G + (size_t)g * BL * H_SZ;
  const int half = lane >> 4, nn = lane & 15;
#pragma unroll
  for (int mt = 0; mt < 2; ++mt)
#pragma unroll
    for (int nt = 0; nt < 2; ++nt) {
      int col = col0 + nt * 16 + nn;
#pragma unroll
      for (int r = 0; r < 8; ++r) {
        int m = row0 + mt * 16 + r + 8 * half;   // C/D layout: VGPR r -> M=r(+8)
        Gg[(size_t)m * H_SZ + col] = f2bf(acc[mt][nt][r]);
      }
    }
}

// ---------------- Kernel: persistent recurrent LSTM scan -------------------
// grid: NWG_REC WGs x 128 threads (4 waves). WG w owns H-columns [32w,32w+32).
// All 4 gates' weight slices (4 x 32 x 1024 bf16 = 256 KB) are staged into LDS
// ONCE (async global->LDS on gfx1250), then re-read from LDS for all 512 steps.
// wave v: mt = v>>1 (batch rows), nt = v&1 (16-col tile); cell state c in regs.
__global__ __launch_bounds__(128) void k_lstm(
    const uint16_t* __restrict__ Whi, const uint16_t* __restrict__ Whf,
    const uint16_t* __restrict__ Whg, const uint16_t* __restrict__ Who,
    const uint16_t* __restrict__ G,
    const float* __restrict__ bii, const float* __restrict__ bhi,
    const float* __restrict__ bif, const float* __restrict__ bhf,
    const float* __restrict__ big, const float* __restrict__ bhg,
    const float* __restrict__ bio, const float* __restrict__ bho,
    uint16_t* __restrict__ h0buf, uint16_t* __restrict__ h1buf,
    float* __restrict__ out_seq, float* __restrict__ out_c,
    unsigned* __restrict__ bar) {
  extern __shared__ char smem[];
  uint16_t* lW = (uint16_t*)smem;               // [4][HC][H_SZ] bf16 = 256 KB

  const int lane = threadIdx.x & 31;
  const int wv = threadIdx.x >> 5;              // 0..3
  const int mt = wv >> 1;                       // batch tile
  const int nt = wv & 1;                        // column tile within HC
  const int cwg = blockIdx.x * HC;
  const int half = lane >> 4, nn = lane & 15;
  const int colLoc = nt * 16 + nn;              // 0..31 local column
  const int col = cwg + colLoc;                 // global H column
  const uint16_t* __restrict__ Wh[4] = {Whi, Whf, Whg, Who};

  // ---- stage recurrent weights into LDS (one time) ----
  // 4 gates * HC rows * (H_SZ*2/16)=128 uint4 per row => 16384 uint4 total
  {
    const int tid = threadIdx.x;
    for (int i = tid; i < 4 * HC * 128; i += 128) {
      const int g = i >> 12;                    // / (HC*128)
      const int rem = i & 4095;
      const int r = rem >> 7;                   // local column 0..31
      const int kq = rem & 127;                 // uint4 within the row
      const uint4* src =
          (const uint4*)(Wh[g] + (size_t)(cwg + r) * H_SZ) + kq;
      uint4* dst = (uint4*)lW + i;
#if USE_ASYNC_LDS
      __builtin_amdgcn_global_load_async_to_lds_b128(
          (__attribute__((address_space(1))) v4i_async*)src,
          (__attribute__((address_space(3))) v4i_async*)dst, 0, 0);
#else
      *dst = *src;
#endif
    }
#if USE_ASYNC_LDS
    __builtin_amdgcn_s_wait_asynccnt(0);
#endif
    __syncthreads();
  }

  // per-lane bias sums (loop-invariant)
  float bsum[4];
  bsum[0] = bii[col] + bhi[col];
  bsum[1] = bif[col] + bhf[col];
  bsum[2] = big[col] + bhg[col];
  bsum[3] = bio[col] + bho[col];

  float cc[8];
#pragma unroll
  for (int r = 0; r < 8; ++r) cc[r] = 0.0f;

  const int arow = mt * 16 + nn;

  for (int t = 0; t < SEQ; ++t) {
    const uint16_t* __restrict__ hrd = (t & 1) ? h1buf : h0buf;
    uint16_t* __restrict__ hwr = (t & 1) ? h0buf : h1buf;

    v8f acc[4] = {};
#pragma unroll 2
    for (int k0 = 0; k0 < H_SZ; k0 += 32) {
      v16bf a = load_frag(hrd, arow, H_SZ, k0, lane);   // h tile from L2
#pragma unroll
      for (int g = 0; g < 4; ++g) {
        v16bf b = load_frag_s(lW + g * (HC * H_SZ), colLoc, H_SZ, k0, lane);
        acc[g] = wmma_bf16(a, b, acc[g]);
      }
    }

#pragma unroll
    for (int r = 0; r < 8; ++r) {
      const int m = mt * 16 + r + 8 * half;             // batch index
      const size_t grow = ((size_t)m * SEQ + t) * H_SZ + col;
      const size_t gsz = (size_t)BL * H_SZ;
      float pi = acc[0][r] + bf2f(G[0 * gsz + grow]) + bsum[0];
      float pf = acc[1][r] + bf2f(G[1 * gsz + grow]) + bsum[1];
      float pg = acc[2][r] + bf2f(G[2 * gsz + grow]) + bsum[2];
      float po = acc[3][r] + bf2f(G[3 * gsz + grow]) + bsum[3];
      float it = sigmoidf_(pi);
      float ft = sigmoidf_(pf);
      float gt = tanhf(pg);
      float ot = sigmoidf_(po);
      float c = ft * cc[r] + it * gt;
      cc[r] = c;
      float h = ot * tanhf(c);
      out_seq[(size_t)m * SEQ * H_SZ + (size_t)t * H_SZ + col] = h;
      hwr[(size_t)m * H_SZ + col] = f2bf(h);
    }

    // -------- grid barrier (monotonic counter, agent scope) --------
    __syncthreads();
    if (threadIdx.x == 0) {
      __threadfence();
      __hip_atomic_fetch_add(bar, 1u, __ATOMIC_ACQ_REL, __HIP_MEMORY_SCOPE_AGENT);
      const unsigned target = (unsigned)(t + 1) * (unsigned)NWG_REC;
      while (__hip_atomic_load(bar, __ATOMIC_ACQUIRE, __HIP_MEMORY_SCOPE_AGENT) < target)
        __builtin_amdgcn_s_sleep(2);
    }
    __syncthreads();
  }

  // final cell state c_T -> d_out
#pragma unroll
  for (int r = 0; r < 8; ++r) {
    const int m = mt * 16 + r + 8 * half;
    out_c[(size_t)m * H_SZ + col] = cc[r];
  }
}

// ---------------- Kernel: projection logits = h_T @ W_p^T + b_p ------------
// grid: P/64 WGs x 128 threads (4 waves)
__global__ __launch_bounds__(128) void k_proj(
    const uint16_t* __restrict__ hfin, const uint16_t* __restrict__ Wp,
    const float* __restrict__ bp, float* __restrict__ logits) {
  const int lane = threadIdx.x & 31;
  const int wv = threadIdx.x >> 5;
  const int mt = wv >> 1;
  const int jb = (wv & 1) * 2;
  const int cwg = blockIdx.x * 64;
  const int half = lane >> 4, nn = lane & 15;

  v8f acc[2] = {};
#pragma unroll 2
  for (int k0 = 0; k0 < H_SZ; k0 += 32) {
    v16bf a = load_frag(hfin, mt * 16 + nn, H_SZ, k0, lane);
#pragma unroll
    for (int j = 0; j < 2; ++j) {
      v16bf b = load_frag(Wp, cwg + (jb + j) * 16 + nn, H_SZ, k0, lane);
      acc[j] = wmma_bf16(a, b, acc[j]);
    }
  }
#pragma unroll
  for (int j = 0; j < 2; ++j) {
    const int col = cwg + (jb + j) * 16 + nn;
#pragma unroll
    for (int r = 0; r < 8; ++r) {
      const int m = mt * 16 + r + 8 * half;
      logits[(size_t)m * P_SZ + col] = acc[j][r] + bp[col];
    }
  }
}

// ---------------- Kernel: row-wise log_softmax -----------------------------
__global__ __launch_bounds__(256) void k_logsoftmax(
    const float* __restrict__ logits, float* __restrict__ out) {
  __shared__ float red[256];
  const int b = blockIdx.x, tid = threadIdx.x;
  const float* __restrict__ row = logits + (size_t)b * P_SZ;

  float m = -3.402823466e38f;
  for (int j = tid; j < P_SZ; j += 256) m = fmaxf(m, row[j]);
  red[tid] = m;
  __syncthreads();
  for (int s = 128; s > 0; s >>= 1) {
    if (tid < s) red[tid] = fmaxf(red[tid], red[tid + s]);
    __syncthreads();
  }
  const float rmax = red[0];
  __syncthreads();

  float sum = 0.0f;
  for (int j = tid; j < P_SZ; j += 256) sum += __expf(row[j] - rmax);
  red[tid] = sum;
  __syncthreads();
  for (int s = 128; s > 0; s >>= 1) {
    if (tid < s) red[tid] += red[tid + s];
    __syncthreads();
  }
  const float lse = rmax + __logf(red[0]);

  for (int j = tid; j < P_SZ; j += 256) out[(size_t)b * P_SZ + j] = row[j] - lse;
}

// ---------------- Host-side launch ------------------------------------------
extern "C" void kernel_launch(void* const* d_in, const int* in_sizes, int n_in,
                              void* d_out, int out_size, void* d_ws, size_t ws_size,
                              hipStream_t stream) {
  const float* x    = (const float*)d_in[0];
  const float* W_ii = (const float*)d_in[1];  const float* b_ii = (const float*)d_in[2];
  const float* W_hi = (const float*)d_in[3];  const float* b_hi = (const float*)d_in[4];
  const float* W_if = (const float*)d_in[5];  const float* b_if = (const float*)d_in[6];
  const float* W_hf = (const float*)d_in[7];  const float* b_hf = (const float*)d_in[8];
  const float* W_ig = (const float*)d_in[9];  const float* b_ig = (const float*)d_in[10];
  const float* W_hg = (const float*)d_in[11]; const float* b_hg = (const float*)d_in[12];
  const float* W_io = (const float*)d_in[13]; const float* b_io = (const float*)d_in[14];
  const float* W_ho = (const float*)d_in[15]; const float* b_ho = (const float*)d_in[16];
  const float* W_p  = (const float*)d_in[17]; const float* b_p  = (const float*)d_in[18];

  char* ws = (char*)d_ws;
  size_t off = 0;
  auto alloc = [&](size_t bytes) -> void* {
    void* p = ws + off;
    off = (off + bytes + 255) & ~(size_t)255;
    return p;
  };
  uint16_t* Xb  = (uint16_t*)alloc((size_t)BL * I_SZ * 2);
  uint16_t* Wi[4], *Wh[4];
  for (int g = 0; g < 4; ++g) Wi[g] = (uint16_t*)alloc((size_t)H_SZ * I_SZ * 2);
  for (int g = 0; g < 4; ++g) Wh[g] = (uint16_t*)alloc((size_t)H_SZ * H_SZ * 2);
  uint16_t* Wpb = (uint16_t*)alloc((size_t)P_SZ * H_SZ * 2);
  uint16_t* Gb  = (uint16_t*)alloc((size_t)4 * BL * H_SZ * 2);
  uint16_t* h0b = (uint16_t*)alloc((size_t)BATCH * H_SZ * 2);
  uint16_t* h1b = (uint16_t*)alloc((size_t)BATCH * H_SZ * 2);
  float*   lgt  = (float*)alloc((size_t)BATCH * P_SZ * 4);
  unsigned* bar = (unsigned*)alloc(256);

  float* out = (float*)d_out;
  float* out_seq  = out;                                  // [B,L,H]
  float* out_logp = out + (size_t)BATCH * SEQ * H_SZ;     // [B,P]
  float* out_c    = out_logp + (size_t)BATCH * P_SZ;      // [B,H]

  // 1) fp32 -> bf16 conversions
  auto conv = [&](const float* s, uint16_t* d, size_t n) {
    int blocks = (int)((n + 255) / 256);
    if (blocks > 2048) blocks = 2048;
    hipLaunchKernelGGL(k_convert, dim3(blocks), dim3(256), 0, stream, s, d, (int)n);
  };
  conv(x, Xb, (size_t)BL * I_SZ);
  conv(W_ii, Wi[0], (size_t)H_SZ * I_SZ); conv(W_if, Wi[1], (size_t)H_SZ * I_SZ);
  conv(W_ig, Wi[2], (size_t)H_SZ * I_SZ); conv(W_io, Wi[3], (size_t)H_SZ * I_SZ);
  conv(W_hi, Wh[0], (size_t)H_SZ * H_SZ); conv(W_hf, Wh[1], (size_t)H_SZ * H_SZ);
  conv(W_hg, Wh[2], (size_t)H_SZ * H_SZ); conv(W_ho, Wh[3], (size_t)H_SZ * H_SZ);
  conv(W_p, Wpb, (size_t)P_SZ * H_SZ);

  // 2) init ping-pong h buffers + barrier counter
  {
    int n = BATCH * H_SZ;
    hipLaunchKernelGGL(k_init, dim3((n + 255) / 256), dim3(256), 0, stream,
                       h0b, h1b, bar, n);
  }

  // 3) input-side GEMMs (full-chip WMMA): G[g] = X @ Wi[g]^T, stored bf16
  hipLaunchKernelGGL(k_in_gemm, dim3(H_SZ / 64, BL / 128, 4), dim3(256), 0, stream,
                     Xb, Wi[0], Wi[1], Wi[2], Wi[3], Gb);

  // 4) persistent recurrent scan (32 WGs, 256KB LDS weight residency,
  //    grid barrier per timestep)
  hipLaunchKernelGGL(k_lstm, dim3(NWG_REC), dim3(128),
                     4 * HC * H_SZ * sizeof(uint16_t), stream,
                     Wh[0], Wh[1], Wh[2], Wh[3], Gb,
                     b_ii, b_hi, b_if, b_hf, b_ig, b_hg, b_io, b_ho,
                     h0b, h1b, out_seq, out_c, bar);

  // 5) projection: logits = h_T @ W_p^T + b_p   (final h lives in h0b: L even)
  hipLaunchKernelGGL(k_proj, dim3(P_SZ / 64), dim3(128), 0, stream,
                     h0b, Wpb, b_p, lgt);

  // 6) log_softmax rows -> out_logp
  hipLaunchKernelGGL(k_logsoftmax, dim3(BATCH), dim3(256), 0, stream,
                     lgt, out_logp);

  (void)in_sizes; (void)n_in; (void)out_size; (void)ws_size;
}
